// Transformer2D_36661840839168
// MI455X (gfx1250) — compile-verified, matrix operand order checked by reading, not run on hardware
//
#include <hip/hip_runtime.h>
#include <math.h>

// ---------------------------------------------------------------------------
// CDNA5 WMMA (wave32): D(16x16 f32) = A(16x32 f16) * B(32x16 f16) + C
// ---------------------------------------------------------------------------
typedef __attribute__((ext_vector_type(16))) _Float16 v16h;
typedef __attribute__((ext_vector_type(8)))  float    v8f;

__device__ __forceinline__ v8f wmma16(v16h a, v16h b, v8f c) {
  return __builtin_amdgcn_wmma_f32_16x16x32_f16(false, a, false, b, (short)0, c,
                                                false, false);
}

// A-fragment (16x32 f16) from LDS [row][k], stride ld.
__device__ __forceinline__ v16h frag_a(const _Float16* src, int ld, int row0,
                                       int k0, int lane) {
  const _Float16* p = src + (size_t)(row0 + (lane & 15)) * ld + k0 + ((lane >> 4) << 3);
  v16h f;
#pragma unroll
  for (int i = 0; i < 8; ++i) {
    int k = ((i >> 2) << 4) + ((i & 3) << 1);
    f[2 * i]     = p[k];
    f[2 * i + 1] = p[k + 1];
  }
  return f;
}

// B-fragment (32x16 f16) from LDS stored TRANSPOSED [n][k], stride ld.
__device__ __forceinline__ v16h frag_b(const _Float16* srcT, int ld, int n0,
                                       int k0, int lane) {
  const _Float16* p = srcT + (size_t)(n0 + (lane & 15)) * ld + k0 + ((lane >> 4) << 3);
  v16h f;
#pragma unroll
  for (int i = 0; i < 8; ++i) {
    int k = ((i >> 2) << 4) + ((i & 3) << 1);
    f[2 * i]     = p[k];
    f[2 * i + 1] = p[k + 1];
  }
  return f;
}

// packed f32x2 -> f16x2 in one dword via v_cvt_pk_rtz_f16_f32
__device__ __forceinline__ unsigned int pk2(float a, float b) {
  return __builtin_bit_cast(unsigned int, __builtin_amdgcn_cvt_pkrtz(a, b));
}
__device__ __forceinline__ uint4 pk8p(const float* p) {
  uint4 u;
  u.x = pk2(p[0], p[1]); u.y = pk2(p[2], p[3]);
  u.z = pk2(p[4], p[5]); u.w = pk2(p[6], p[7]);
  return u;
}

#define LDK 40   // LDS halfword stride: 80B (16B multiple)

// ---------------------------------------------------------------------------
// Generic GEMM: C[M,N] = op( A[M,K] * W[K,N] + bias )
// MODE 0: store   MODE 1: C +=   MODE 2: final ([B,S,C]->[B,C,H,W] + resq)
// GUARD: row guard for partial M tiles (only cross-K/V needs it).
// 256 thr / 8 waves, tile 128x128, K-step 32, double-buffered LDS.
// ---------------------------------------------------------------------------
template <int MODE, bool GUARD>
__global__ __launch_bounds__(256) void gemm_kernel(
    const float* __restrict__ A, const float* __restrict__ W,
    const float* __restrict__ bias, float* __restrict__ C,
    const float* __restrict__ resq, int M, int N, int K) {
  __shared__ _Float16 Ash[2][128][LDK];   // [m][k]
  __shared__ _Float16 BshT[2][128][LDK];  // [n][k]

  int t = threadIdx.x, wave = t >> 5, lane = t & 31;
  int bm = blockIdx.y * 128, bn = blockIdx.x * 128;
  int wm = (wave >> 1) * 32;
  int wn = (wave & 1) * 64;

  // A staging: thread -> (row, 16 contiguous k); branchless M-guard via clamp
  int arow = t >> 1, ac0 = (t & 1) * 16;
  bool inm = GUARD ? ((bm + arow) < M) : true;
  const float* aptr = A + (size_t)(inm ? (bm + arow) : 0) * K + ac0;
  // B staging: thread -> (col n, 16-k strip), coalesced across lanes
  int bnn = t & 127, bkh = (t >> 7) * 16;
  const float* wptr = W + (size_t)bkh * N + bn + bnn;

  float arf[16], brf[16];
  auto loadA = [&](int kb) {
#pragma unroll
    for (int i = 0; i < 4; ++i) {
      float4 v = *(const float4*)(aptr + kb + 4 * i);
      if (GUARD && !inm) { v.x = 0.f; v.y = 0.f; v.z = 0.f; v.w = 0.f; }
      arf[4 * i] = v.x; arf[4 * i + 1] = v.y;
      arf[4 * i + 2] = v.z; arf[4 * i + 3] = v.w;
    }
  };
  auto loadB = [&](int kb) {
#pragma unroll
    for (int j = 0; j < 16; ++j) brf[j] = wptr[(size_t)(kb + j) * N];
  };
  auto storeA = [&](int buf) {
    *(uint4*)&Ash[buf][arow][ac0]     = pk8p(arf);
    *(uint4*)&Ash[buf][arow][ac0 + 8] = pk8p(arf + 8);
  };
  auto storeB = [&](int buf) {
    *(uint4*)&BshT[buf][bnn][bkh]     = pk8p(brf);
    *(uint4*)&BshT[buf][bnn][bkh + 8] = pk8p(brf + 8);
  };

  v8f acc[2][4] = {};

  loadA(0); loadB(0);
  storeA(0); storeB(0);
  __syncthreads();

  int iters = K >> 5;
  for (int it = 0; it < iters; ++it) {
    int cur = it & 1;
    bool more = (it + 1) < iters;
    if (more) { loadA((it + 1) << 5); loadB((it + 1) << 5); }

    // hoist all fragments first -> batched ds_loads, partial dscnt waits
    v16h af[2], bf[4];
    af[0] = frag_a(&Ash[cur][0][0], LDK, wm, 0, lane);
    af[1] = frag_a(&Ash[cur][0][0], LDK, wm + 16, 0, lane);
#pragma unroll
    for (int nt = 0; nt < 4; ++nt)
      bf[nt] = frag_b(&BshT[cur][0][0], LDK, wn + nt * 16, 0, lane);
#pragma unroll
    for (int nt = 0; nt < 4; ++nt) {
      acc[0][nt] = wmma16(af[0], bf[nt], acc[0][nt]);
      acc[1][nt] = wmma16(af[1], bf[nt], acc[1][nt]);
    }
    if (more) { storeA(cur ^ 1); storeB(cur ^ 1); }
    __syncthreads();
  }

  int nlo = lane & 15, half = lane >> 4;
#pragma unroll
  for (int mi = 0; mi < 2; ++mi) {
#pragma unroll
    for (int nt = 0; nt < 4; ++nt) {
      int col = bn + wn + nt * 16 + nlo;
      float bv = bias ? bias[col] : 0.f;
#pragma unroll
      for (int v = 0; v < 8; ++v) {
        int row = bm + wm + mi * 16 + v + 8 * half;
        if (!GUARD || row < M) {
          float val = acc[mi][nt][v] + bv;
          if (MODE == 0) {
            C[(size_t)row * N + col] = val;
          } else if (MODE == 1) {
            C[(size_t)row * N + col] += val;
          } else {
            int b = row >> 10, s = row & 1023;
            size_t dst = (((size_t)b * 512 + col) << 10) + s;
            C[dst] = val + resq[dst];
          }
        }
      }
    }
  }
}

// ---------------------------------------------------------------------------
// GEGLU GEMM: out[M,2048] = (A*Wa + ba) * gelu(A*Wg + bg)
// A[M,512], W[512,4096]. 256 thr, tile 128x64, dual accumulators, dbl-buffer.
// ---------------------------------------------------------------------------
__global__ __launch_bounds__(256) void gemm_geglu_kernel(
    const float* __restrict__ A, const float* __restrict__ W,
    const float* __restrict__ bias, float* __restrict__ C) {
  const int K = 512, NT = 4096, NO = 2048;
  __shared__ _Float16 Ash[2][128][LDK];
  __shared__ _Float16 BaT[2][64][LDK];
  __shared__ _Float16 BgT[2][64][LDK];

  int t = threadIdx.x, wave = t >> 5, lane = t & 31;
  int bm = blockIdx.y * 128, bn = blockIdx.x * 64;

  int arow = t >> 1, ac0 = (t & 1) * 16;
  const float* aptr = A + (size_t)(bm + arow) * K + ac0;
  int bnn = t & 63, bkh = (t >> 6) * 8;
  const float* waptr = W + (size_t)bkh * NT + bn + bnn;
  const float* wgptr = waptr + NO;

  float arf[16], baf[8], bgf[8];
  auto loadA = [&](int kb) {
#pragma unroll
    for (int i = 0; i < 4; ++i) {
      float4 v = *(const float4*)(aptr + kb + 4 * i);
      arf[4 * i] = v.x; arf[4 * i + 1] = v.y;
      arf[4 * i + 2] = v.z; arf[4 * i + 3] = v.w;
    }
  };
  auto loadB = [&](int kb) {
#pragma unroll
    for (int j = 0; j < 8; ++j) {
      baf[j] = waptr[(size_t)(kb + j) * NT];
      bgf[j] = wgptr[(size_t)(kb + j) * NT];
    }
  };
  auto storeA = [&](int buf) {
    *(uint4*)&Ash[buf][arow][ac0]     = pk8p(arf);
    *(uint4*)&Ash[buf][arow][ac0 + 8] = pk8p(arf + 8);
  };
  auto storeB = [&](int buf) {
    *(uint4*)&BaT[buf][bnn][bkh] = pk8p(baf);
    *(uint4*)&BgT[buf][bnn][bkh] = pk8p(bgf);
  };

  v8f acc_a[4] = {};
  v8f acc_g[4] = {};

  loadA(0); loadB(0);
  storeA(0); storeB(0);
  __syncthreads();

  const int iters = K >> 5;
  for (int it = 0; it < iters; ++it) {
    int cur = it & 1;
    bool more = (it + 1) < iters;
    if (more) { loadA((it + 1) << 5); loadB((it + 1) << 5); }

    v16h af = frag_a(&Ash[cur][0][0], LDK, wave * 16, 0, lane);
    v16h ba[4], bg[4];
#pragma unroll
    for (int nt = 0; nt < 4; ++nt) {
      ba[nt] = frag_b(&BaT[cur][0][0], LDK, nt * 16, 0, lane);
      bg[nt] = frag_b(&BgT[cur][0][0], LDK, nt * 16, 0, lane);
    }
#pragma unroll
    for (int nt = 0; nt < 4; ++nt) {
      acc_a[nt] = wmma16(af, ba[nt], acc_a[nt]);
      acc_g[nt] = wmma16(af, bg[nt], acc_g[nt]);
    }
    if (more) { storeA(cur ^ 1); storeB(cur ^ 1); }
    __syncthreads();
  }

  int nlo = lane & 15, half = lane >> 4;
#pragma unroll
  for (int nt = 0; nt < 4; ++nt) {
    int col = bn + nt * 16 + nlo;
    float bva = bias[col], bvg = bias[col + NO];
#pragma unroll
    for (int v = 0; v < 8; ++v) {
      int row = bm + wave * 16 + v + 8 * half;
      float a = acc_a[nt][v] + bva;
      float g = acc_g[nt][v] + bvg;
      float gl = 0.5f * g * (1.0f + erff(g * 0.70710678118f));
      C[(size_t)row * NO + col] = a * gl;
    }
  }
}

// ---------------------------------------------------------------------------
// Flash attention (8 heads, dim 64, Sq=1024). Q/K/V: [rows,512], head at h*64.
// Sk = 1024 (self) or 77 (cross). 128 thr / 4 waves, q-tile 64 (16/wave),
// key chunks of 64, WMMA for QK^T and PV, register prefetch of next chunk.
// ---------------------------------------------------------------------------
#define LDA 72   // attention LDS halfword stride (144B, 16B multiple)
__global__ __launch_bounds__(128) void attn_kernel(
    const float* __restrict__ Qg, const float* __restrict__ Kg,
    const float* __restrict__ Vg, float* __restrict__ Og, int Sk, float scale) {
  const int D = 64, LD = 512, SQ = 1024;
  __shared__ _Float16 Qs[64][LDA];      // [qrow][dim]
  __shared__ _Float16 Ks[64][LDA];      // [key][dim]
  __shared__ _Float16 VsT[64][LDA];     // [dim][key]
  __shared__ _Float16 Ps[4][16][LDA];   // per-wave P tile [qrow][key]

  int t = threadIdx.x, wave = t >> 5, lane = t & 31;
  int qt = blockIdx.x, head = blockIdx.y, b = blockIdx.z;
  int nlo = lane & 15, half = lane >> 4;

  size_t qbase = ((size_t)b * SQ + qt * 64) * LD + head * D;
  size_t kvbase = (size_t)b * Sk * LD + head * D;

  // ---- stage Q tile 64x64 (once) ----
  {
    int row = t >> 1, c0 = (t & 1) * 32;
    const float* qp = Qg + qbase + (size_t)row * LD + c0;
    float qf[32];
#pragma unroll
    for (int i = 0; i < 8; ++i) {
      float4 v = *(const float4*)(qp + 4 * i);
      qf[4 * i] = v.x; qf[4 * i + 1] = v.y; qf[4 * i + 2] = v.z; qf[4 * i + 3] = v.w;
    }
#pragma unroll
    for (int i = 0; i < 4; ++i)
      *(uint4*)&Qs[row][c0 + 8 * i] = pk8p(qf + 8 * i);
  }

  // K: thread -> (key row, 32 dims); V: thread -> (dim col, 32-key strip)
  int krow = t >> 1, kc0 = (t & 1) * 32;
  int vdim = t & 63, vkh = (t >> 6) * 32;
  float kf[32], vf[32];

  auto loadKV = [&](int kc) {
    int key = kc + krow;
    bool ink = key < Sk;
    const float* kp = Kg + kvbase + (size_t)(ink ? key : 0) * LD + kc0;
#pragma unroll
    for (int i = 0; i < 8; ++i) {
      float4 v = *(const float4*)(kp + 4 * i);
      if (!ink) { v.x = 0.f; v.y = 0.f; v.z = 0.f; v.w = 0.f; }
      kf[4 * i] = v.x; kf[4 * i + 1] = v.y; kf[4 * i + 2] = v.z; kf[4 * i + 3] = v.w;
    }
#pragma unroll
    for (int j = 0; j < 32; ++j) {
      int vkey = kc + vkh + j;
      bool inv = vkey < Sk;
      float v = Vg[kvbase + (size_t)(inv ? vkey : 0) * LD + vdim];
      vf[j] = inv ? v : 0.f;
    }
  };
  auto storeKV = [&]() {
#pragma unroll
    for (int i = 0; i < 4; ++i) {
      *(uint4*)&Ks[krow][kc0 + 8 * i]  = pk8p(kf + 8 * i);
      *(uint4*)&VsT[vdim][vkh + 8 * i] = pk8p(vf + 8 * i);
    }
  };

  float m_i[8], l_i[8];
#pragma unroll
  for (int v = 0; v < 8; ++v) { m_i[v] = -1e30f; l_i[v] = 0.f; }
  v8f acc_o[4] = {};

  loadKV(0);
  for (int kc = 0; kc < Sk; kc += 64) {
    storeKV();
    __syncthreads();
    if (kc + 64 < Sk) loadKV(kc + 64);

    // S = Q * K^T   (hoist fragments for batched ds_loads)
    v8f s_acc[4] = {};
#pragma unroll
    for (int ks = 0; ks < 64; ks += 32) {
      v16h qa = frag_a(&Qs[0][0], LDA, wave * 16, ks, lane);
      v16h kb_[4];
#pragma unroll
      for (int nt = 0; nt < 4; ++nt)
        kb_[nt] = frag_b(&Ks[0][0], LDA, nt * 16, ks, lane);
#pragma unroll
      for (int nt = 0; nt < 4; ++nt)
        s_acc[nt] = wmma16(qa, kb_[nt], s_acc[nt]);
    }

    // scale + mask + online softmax
    float newm[8];
#pragma unroll
    for (int v = 0; v < 8; ++v) newm[v] = m_i[v];
#pragma unroll
    for (int nt = 0; nt < 4; ++nt) {
      bool valid = (kc + nt * 16 + nlo) < Sk;
#pragma unroll
      for (int v = 0; v < 8; ++v) {
        float sv = s_acc[nt][v] * scale;
        if (!valid) sv = -1e30f;
        s_acc[nt][v] = sv;
        newm[v] = fmaxf(newm[v], sv);
      }
    }
#pragma unroll
    for (int mask = 1; mask <= 8; mask <<= 1)
#pragma unroll
      for (int v = 0; v < 8; ++v)
        newm[v] = fmaxf(newm[v], __shfl_xor(newm[v], mask, 32));

    float alpha[8], rsum[8];
#pragma unroll
    for (int v = 0; v < 8; ++v) { alpha[v] = __expf(m_i[v] - newm[v]); rsum[v] = 0.f; }
#pragma unroll
    for (int nt = 0; nt < 4; ++nt)
#pragma unroll
      for (int v = 0; v < 8; ++v) {
        float p = __expf(s_acc[nt][v] - newm[v]);
        s_acc[nt][v] = p;
        rsum[v] += p;
      }
#pragma unroll
    for (int mask = 1; mask <= 8; mask <<= 1)
#pragma unroll
      for (int v = 0; v < 8; ++v) rsum[v] += __shfl_xor(rsum[v], mask, 32);
#pragma unroll
    for (int v = 0; v < 8; ++v) {
      l_i[v] = l_i[v] * alpha[v] + rsum[v];
      m_i[v] = newm[v];
    }
#pragma unroll
    for (int nt = 0; nt < 4; ++nt)
#pragma unroll
      for (int v = 0; v < 8; ++v) acc_o[nt][v] *= alpha[v];

    // P: C-layout -> per-wave LDS -> A-layout
#pragma unroll
    for (int nt = 0; nt < 4; ++nt)
#pragma unroll
      for (int v = 0; v < 8; ++v)
        Ps[wave][v + 8 * half][nt * 16 + nlo] = (_Float16)s_acc[nt][v];
    asm volatile("s_wait_dscnt 0x0" ::: "memory");

#pragma unroll
    for (int ks = 0; ks < 64; ks += 32) {
      v16h pa = frag_a(&Ps[wave][0][0], LDA, 0, ks, lane);
      v16h vb[4];
#pragma unroll
      for (int nt = 0; nt < 4; ++nt)
        vb[nt] = frag_b(&VsT[0][0], LDA, nt * 16, ks, lane);
#pragma unroll
      for (int nt = 0; nt < 4; ++nt)
        acc_o[nt] = wmma16(pa, vb[nt], acc_o[nt]);
    }
    __syncthreads();
  }

#pragma unroll
  for (int nt = 0; nt < 4; ++nt)
#pragma unroll
    for (int v = 0; v < 8; ++v) {
      size_t dst = qbase + (size_t)(wave * 16 + v + 8 * half) * LD + nt * 16 + nlo;
      Og[dst] = acc_o[nt][v] / l_i[v];
    }
}

// ---------------------------------------------------------------------------
// GroupNorm on q [8,512,32,32], 32 groups -> transposed [B, S=1024, C=512]
// ---------------------------------------------------------------------------
__global__ __launch_bounds__(256) void gn_kernel(
    const float* __restrict__ q, const float* __restrict__ g,
    const float* __restrict__ beta, float* __restrict__ out) {
  int bg = blockIdx.x, b = bg >> 5, grp = bg & 31;
  size_t base = ((size_t)b * 512 + grp * 16) << 10;
  __shared__ float ssum[256], ssq[256];

  float s = 0.f, s2 = 0.f;
  for (int e = threadIdx.x; e < 16 * 1024; e += 256) {
    float v = q[base + e];
    s += v; s2 += v * v;
  }
  ssum[threadIdx.x] = s; ssq[threadIdx.x] = s2;
  __syncthreads();
  for (int o = 128; o > 0; o >>= 1) {
    if (threadIdx.x < o) {
      ssum[threadIdx.x] += ssum[threadIdx.x + o];
      ssq[threadIdx.x] += ssq[threadIdx.x + o];
    }
    __syncthreads();
  }
  float mean = ssum[0] * (1.f / 16384.f);
  float var = ssq[0] * (1.f / 16384.f) - mean * mean;
  float rs = rsqrtf(var + 1e-6f);

  for (int e = threadIdx.x; e < 16 * 1024; e += 256) {
    int ch = e >> 10, sp = e & 1023;
    int c = grp * 16 + ch;
    float v = (q[base + e] - mean) * rs * g[c] + beta[c];
    out[(((size_t)b << 10) + sp) * 512 + c] = v;
  }
}

// ---------------------------------------------------------------------------
// LayerNorm rows of [8192, 512]
// ---------------------------------------------------------------------------
__global__ __launch_bounds__(256) void ln_kernel(
    const float* __restrict__ x, const float* __restrict__ g,
    const float* __restrict__ b, float* __restrict__ y) {
  const int C = 512;
  size_t row = blockIdx.x;
  const float* xr = x + row * C;
  __shared__ float ssum[256], ssq[256];
  float v0 = xr[threadIdx.x], v1 = xr[threadIdx.x + 256];
  ssum[threadIdx.x] = v0 + v1;
  ssq[threadIdx.x] = v0 * v0 + v1 * v1;
  __syncthreads();
  for (int o = 128; o > 0; o >>= 1) {
    if (threadIdx.x < o) {
      ssum[threadIdx.x] += ssum[threadIdx.x + o];
      ssq[threadIdx.x] += ssq[threadIdx.x + o];
    }
    __syncthreads();
  }
  float mean = ssum[0] * (1.f / 512.f);
  float var = ssq[0] * (1.f / 512.f) - mean * mean;
  float rs = rsqrtf(var + 1e-5f);
  int i0 = threadIdx.x, i1 = threadIdx.x + 256;
  y[row * C + i0] = (v0 - mean) * rs * g[i0] + b[i0];
  y[row * C + i1] = (v1 - mean) * rs * g[i1] + b[i1];
}

// ---------------------------------------------------------------------------
extern "C" void kernel_launch(void* const* d_in, const int* in_sizes, int n_in,
                              void* d_out, int out_size, void* d_ws, size_t ws_size,
                              hipStream_t stream) {
  const float* q       = (const float*)d_in[0];
  const float* kv      = (const float*)d_in[1];
  const float* gn_g    = (const float*)d_in[2];
  const float* gn_b    = (const float*)d_in[3];
  const float* ln2_g   = (const float*)d_in[4];
  const float* ln2_b   = (const float*)d_in[5];
  const float* ln3_g   = (const float*)d_in[6];
  const float* ln3_b   = (const float*)d_in[7];
  const float* ln4_g   = (const float*)d_in[8];
  const float* ln4_b   = (const float*)d_in[9];
  const float* fc_in_w = (const float*)d_in[10];
  const float* fc_in_b = (const float*)d_in[11];
  const float* fc_out_w= (const float*)d_in[12];
  const float* fc_out_b= (const float*)d_in[13];
  const float* a1_q    = (const float*)d_in[14];
  const float* a1_k    = (const float*)d_in[15];
  const float* a1_v    = (const float*)d_in[16];
  const float* a1_o    = (const float*)d_in[17];
  const float* a1_ob   = (const float*)d_in[18];
  const float* a2_q    = (const float*)d_in[19];
  const float* a2_k    = (const float*)d_in[20];
  const float* a2_v    = (const float*)d_in[21];
  const float* a2_o    = (const float*)d_in[22];
  const float* a2_ob   = (const float*)d_in[23];
  const float* ffp_w   = (const float*)d_in[24];
  const float* ffp_b   = (const float*)d_in[25];
  const float* ffo_w   = (const float*)d_in[26];
  const float* ffo_b   = (const float*)d_in[27];
  float* out = (float*)d_out;

  const int M = 8192;
  const size_t SC = (size_t)M * 512;
  float* X  = (float*)d_ws;    // running activation [8192,512]
  float* T  = X + SC;          // norm / attn output [8192,512]
  float* S3 = T + SC;          // [8192,2048]: Qb|Kb|Vb, later GEGLU out
  float* Qb = S3;
  float* Kb = S3 + SC;
  float* Vb = S3 + 2 * SC;
  float* G  = S3;
  float* CK = S3 + 4 * SC;     // cross K [616,512]
  float* CV = CK + (size_t)640 * 512;

  dim3 blk256(256), blk128(128);
  dim3 gemm_g(4, 64);
  dim3 gemm_kv(4, 5);
  dim3 geglu_g(32, 64);
  dim3 attn_g(16, 8, 8);

  gn_kernel<<<dim3(256), blk256, 0, stream>>>(q, gn_g, gn_b, T);
  gemm_kernel<0, false><<<gemm_g, blk256, 0, stream>>>(T, fc_in_w, fc_in_b, X, nullptr, M, 512, 512);
  ln_kernel<<<dim3(M), blk256, 0, stream>>>(X, ln2_g, ln2_b, T);
  gemm_kernel<0, false><<<gemm_g, blk256, 0, stream>>>(T, a1_q, nullptr, Qb, nullptr, M, 512, 512);
  gemm_kernel<0, false><<<gemm_g, blk256, 0, stream>>>(T, a1_k, nullptr, Kb, nullptr, M, 512, 512);
  gemm_kernel<0, false><<<gemm_g, blk256, 0, stream>>>(T, a1_v, nullptr, Vb, nullptr, M, 512, 512);
  attn_kernel<<<attn_g, blk128, 0, stream>>>(Qb, Kb, Vb, T, 1024, 0.125f);
  gemm_kernel<1, false><<<gemm_g, blk256, 0, stream>>>(T, a1_o, a1_ob, X, nullptr, M, 512, 512);
  ln_kernel<<<dim3(M), blk256, 0, stream>>>(X, ln3_g, ln3_b, T);
  gemm_kernel<0, false><<<gemm_g, blk256, 0, stream>>>(T, a2_q, nullptr, Qb, nullptr, M, 512, 512);
  gemm_kernel<0, true><<<gemm_kv, blk256, 0, stream>>>(kv, a2_k, nullptr, CK, nullptr, 616, 512, 1024);
  gemm_kernel<0, true><<<gemm_kv, blk256, 0, stream>>>(kv, a2_v, nullptr, CV, nullptr, 616, 512, 1024);
  attn_kernel<<<attn_g, blk128, 0, stream>>>(Qb, CK, CV, T, 77, 0.125f);
  gemm_kernel<1, false><<<gemm_g, blk256, 0, stream>>>(T, a2_o, a2_ob, X, nullptr, M, 512, 512);
  ln_kernel<<<dim3(M), blk256, 0, stream>>>(X, ln4_g, ln4_b, T);
  gemm_geglu_kernel<<<geglu_g, blk256, 0, stream>>>(T, ffp_w, ffp_b, G);
  gemm_kernel<1, false><<<gemm_g, blk256, 0, stream>>>(G, ffo_w, ffo_b, X, nullptr, M, 512, 2048);
  gemm_kernel<2, false><<<gemm_g, blk256, 0, stream>>>(X, fc_out_w, fc_out_b, out, q, M, 512, 512);
}
